// Encoder_18700287607402
// MI455X (gfx1250) — matrix operand order repaired and checked by reading, not compiled, not run on hardware
//
#include <hip/hip_runtime.h>
#include <hip/hip_bf16.h>

#define SEQ    512
#define BATCH  64
#define IN_DIM 512
#define HID    1024
#define G3     (3 * HID)   // 3072

typedef __bf16 bf16_t;
typedef __attribute__((ext_vector_type(16))) __bf16 v16bf;
typedef __attribute__((ext_vector_type(8)))  __bf16 v8bf;
typedef __attribute__((ext_vector_type(8)))  float  v8f;

// ---------------------------------------------------------------------------
// f32 -> bf16 conversion (embedding / activations)
// ---------------------------------------------------------------------------
__global__ void f32_to_bf16_kernel(const float* __restrict__ in,
                                   bf16_t* __restrict__ out, int n) {
  int i = blockIdx.x * blockDim.x + threadIdx.x;
  if (i < n) out[i] = (bf16_t)in[i];
}

// ---------------------------------------------------------------------------
// Pack weight W (N x K row-major f32) into WMMA B-fragment order (bf16).
// B = W^T (K x N). Tile (kt, nt) covers K rows [kt*32, kt*32+32),
// N cols [nt*16, nt*16+16). Within a tile: lane -> N col (lane&15),
// 16 contiguous bf16 per lane = K rows in table order: s<8 -> 8*(lane>>4)+s, s>=8 -> 16+8*(lane>>4)+(s-8).
// Tile order: tile = nt * (K/32) + kt  (K-consecutive tiles contiguous).
// ---------------------------------------------------------------------------
__global__ void pack_b_kernel(const float* __restrict__ W,
                              bf16_t* __restrict__ out, int N, int K) {
  int i = blockIdx.x * blockDim.x + threadIdx.x;
  if (i >= N * K) return;
  int tile   = i >> 9;        // 512 elements per tile
  int within = i & 511;
  int lane   = within >> 4;   // 0..31  -> N col (lane&15), K half (lane>>4)
  int h      = within & 15;   // 0..15  -> K element slot in lane
  int Ktiles = K >> 5;
  int nt = tile / Ktiles;
  int kt = tile - nt * Ktiles;
  out[i] = (bf16_t)W[(size_t)(nt * 16 + (lane & 15)) * K + (kt * 32 + (h & 7) + ((lane >> 4) << 3) + ((h >> 3) << 4))];
}

// ---------------------------------------------------------------------------
// Zero-init hidden state (f32 working copy + bf16 matmul copy)
// ---------------------------------------------------------------------------
__global__ void init_h_kernel(float* __restrict__ hf, bf16_t* __restrict__ hb) {
  int i = blockIdx.x * blockDim.x + threadIdx.x;
  if (i < BATCH * HID) { hf[i] = 0.0f; hb[i] = (bf16_t)0.0f; }
}

// ---------------------------------------------------------------------------
// WMMA GEMM:  C[M x N] = A[M x K] (bf16, row-major) * B (pre-packed bf16)
//             + bias[N]  (f32 accumulate, f32 out)
// Block: 256 threads = 8 waves. Each wave computes a 16 x 64 strip:
//   mt = blockIdx.y ; nt0 = (blockIdx.x*8 + wave)*4 (four 16x16 N tiles).
// Per K-step (K tile of 32): one A-frag load reused by 4 WMMAs.
// M % 16 == 0, N % 512 == 0, K % 32 == 0 (holds for all call sites).
// ---------------------------------------------------------------------------
__global__ __launch_bounds__(256)
void gemm_bf16_wmma_kernel(const bf16_t* __restrict__ A,
                           const bf16_t* __restrict__ Bp,
                           const float*  __restrict__ bias,
                           float* __restrict__ C,
                           int M, int N, int K) {
  const int lane   = threadIdx.x & 31;
  const int wave   = threadIdx.x >> 5;
  const int Ktiles = K >> 5;
  const int mt  = blockIdx.y;
  const int nt0 = (blockIdx.x * 8 + wave) * 4;
  const int m0  = mt * 16;

  // A-frag addressing per CDNA5 16-bit A layout:
  // lanes 0-15  (row m0+lane):    halves 0-7 = K+0..7,  halves 8-15 = K+16..23
  // lanes 16-31 (row m0+lane-16): halves 0-7 = K+8..15, halves 8-15 = K+24..31
  const int arow = m0 + (lane & 15);
  const int kb   = (lane >> 4) * 8;
  const bf16_t* Arow = A + (size_t)arow * K;

  const size_t tstride = (size_t)Ktiles * 512; // elements between consecutive nt tiles
  const bf16_t* Bbase = Bp + (size_t)nt0 * tstride + lane * 16;

  v8f acc[4] = {v8f{}, v8f{}, v8f{}, v8f{}};

  for (int kt = 0; kt < Ktiles; ++kt) {
    v8bf alo = *(const v8bf*)(Arow + kt * 32 + kb);
    v8bf ahi = *(const v8bf*)(Arow + kt * 32 + kb + 16);
    v16bf a = __builtin_shufflevector(alo, ahi,
                                      0, 1, 2, 3, 4, 5, 6, 7,
                                      8, 9, 10, 11, 12, 13, 14, 15);
    const bf16_t* bk = Bbase + (size_t)kt * 512;
    v16bf b0 = *(const v16bf*)(bk);
    v16bf b1 = *(const v16bf*)(bk + tstride);
    v16bf b2 = *(const v16bf*)(bk + 2 * tstride);
    v16bf b3 = *(const v16bf*)(bk + 3 * tstride);
    acc[0] = __builtin_amdgcn_wmma_f32_16x16x32_bf16(false, a, false, b0,
                                                     (short)0, acc[0], false, false);
    acc[1] = __builtin_amdgcn_wmma_f32_16x16x32_bf16(false, a, false, b1,
                                                     (short)0, acc[1], false, false);
    acc[2] = __builtin_amdgcn_wmma_f32_16x16x32_bf16(false, a, false, b2,
                                                     (short)0, acc[2], false, false);
    acc[3] = __builtin_amdgcn_wmma_f32_16x16x32_bf16(false, a, false, b3,
                                                     (short)0, acc[3], false, false);
  }

  // C/D layout: VGPR r holds (row m0+r, col=lane) for lanes 0-15 and
  // (row m0+8+r, col=lane-16) for lanes 16-31.
  const int nn    = lane & 15;
  const int rbase = m0 + (lane >> 4) * 8;
#pragma unroll
  for (int t = 0; t < 4; ++t) {
    const int col = (nt0 + t) * 16 + nn;
    const float bv = bias[col];
#pragma unroll
    for (int r = 0; r < 8; ++r) {
      C[(size_t)(rbase + r) * N + col] = acc[t][r] + bv;
    }
  }
}

// ---------------------------------------------------------------------------
// GRU gate fusion for one timestep s:
//   r = sigmoid(gi_r + gh_r); z = sigmoid(gi_z + gh_z)
//   n = tanh(gi_n + r * gh_n); h' = (1-z)*n + z*h
// Writes h' (f32 state, bf16 matmul copy, output slice [b, s, :]).
// ---------------------------------------------------------------------------
__global__ void gru_gate_kernel(const float* __restrict__ gi,
                                const float* __restrict__ gh,
                                float* __restrict__ hf,
                                bf16_t* __restrict__ hb,
                                float* __restrict__ out,
                                float* __restrict__ out_hidden,
                                int s) {
  int i = blockIdx.x * blockDim.x + threadIdx.x;
  if (i >= BATCH * HID) return;
  const int b = i >> 10;         // HID == 1024
  const int j = i & (HID - 1);

  const float* gis = gi + ((size_t)s * BATCH + b) * G3;
  const float* ghb = gh + (size_t)b * G3;

  const float ir = gis[j];
  const float iz = gis[HID + j];
  const float in_ = gis[2 * HID + j];
  const float hr = ghb[j];
  const float hz = ghb[HID + j];
  const float hn = ghb[2 * HID + j];
  const float h  = hf[i];

  const float r = 1.0f / (1.0f + __expf(-(ir + hr)));
  const float z = 1.0f / (1.0f + __expf(-(iz + hz)));
  const float n = tanhf(in_ + r * hn);
  const float hnew = (1.0f - z) * n + z * h;

  hf[i] = hnew;
  hb[i] = (bf16_t)hnew;
  out[(size_t)b * (SEQ * HID) + (size_t)s * HID + j] = hnew;
  if (s == SEQ - 1) out_hidden[i] = hnew;
}

// ---------------------------------------------------------------------------
extern "C" void kernel_launch(void* const* d_in, const int* in_sizes, int n_in,
                              void* d_out, int out_size, void* d_ws, size_t ws_size,
                              hipStream_t stream) {
  (void)in_sizes; (void)n_in; (void)out_size; (void)ws_size;

  const float* emb  = (const float*)d_in[0];  // (SEQ, BATCH, IN_DIM)
  const float* w_ih = (const float*)d_in[1];  // (3H, IN_DIM)
  const float* w_hh = (const float*)d_in[2];  // (3H, HID)
  const float* b_ih = (const float*)d_in[3];  // (3H,)
  const float* b_hh = (const float*)d_in[4];  // (3H,)

  float* out        = (float*)d_out;                        // (BATCH, SEQ, HID)
  float* out_hidden = out + (size_t)BATCH * SEQ * HID;      // (1, BATCH, HID)

  // Workspace carve-up (256 B aligned slabs)
  char* ws = (char*)d_ws;
  size_t off = 0;
  auto carve = [&](size_t bytes) -> char* {
    char* p = ws + off;
    off = (off + bytes + 255) & ~((size_t)255);
    return p;
  };
  float*  gi    = (float*) carve((size_t)SEQ * BATCH * G3 * sizeof(float)); // 403 MB
  float*  gh    = (float*) carve((size_t)BATCH * G3 * sizeof(float));
  float*  hf    = (float*) carve((size_t)BATCH * HID * sizeof(float));
  bf16_t* hb    = (bf16_t*)carve((size_t)BATCH * HID * sizeof(bf16_t));
  bf16_t* embbf = (bf16_t*)carve((size_t)SEQ * BATCH * IN_DIM * sizeof(bf16_t));
  bf16_t* wihp  = (bf16_t*)carve((size_t)G3 * IN_DIM * sizeof(bf16_t));
  bf16_t* whhp  = (bf16_t*)carve((size_t)G3 * HID * sizeof(bf16_t));

  // 1) One-time conversions / packing
  const int nEmb = SEQ * BATCH * IN_DIM;
  f32_to_bf16_kernel<<<(nEmb + 255) / 256, 256, 0, stream>>>(emb, embbf, nEmb);
  pack_b_kernel<<<(G3 * IN_DIM + 255) / 256, 256, 0, stream>>>(w_ih, wihp, G3, IN_DIM);
  pack_b_kernel<<<(G3 * HID   + 255) / 256, 256, 0, stream>>>(w_hh, whhp, G3, HID);
  init_h_kernel<<<(BATCH * HID + 255) / 256, 256, 0, stream>>>(hf, hb);

  // 2) Input projection: gi[s*B+b, :] = emb @ w_ih^T + b_ih
  {
    dim3 grid(G3 / 512, (SEQ * BATCH) / 16);   // (6, 2048)
    gemm_bf16_wmma_kernel<<<grid, 256, 0, stream>>>(embbf, wihp, b_ih, gi,
                                                    SEQ * BATCH, G3, IN_DIM);
  }

  // 3) Sequential scan: per step, gh = h @ w_hh^T + b_hh, then fused gates.
  {
    dim3 grid(G3 / 512, BATCH / 16);           // (6, 4)
    for (int s = 0; s < SEQ; ++s) {
      gemm_bf16_wmma_kernel<<<grid, 256, 0, stream>>>(hb, whhp, b_hh, gh,
                                                      BATCH, G3, HID);
      gru_gate_kernel<<<(BATCH * HID + 255) / 256, 256, 0, stream>>>(
          gi, gh, hf, hb, out, out_hidden, s);
    }
  }
}
